// EMLResponsibilityPropagation_79663053406676
// MI455X (gfx1250) — matrix operand order repaired and checked by reading, not compiled, not run on hardware
//
#include <hip/hip_runtime.h>
#include <math.h>

// ---------------- problem constants ----------------
constexpr int CB  = 8;     // batch
constexpr int CS  = 4096;  // sequence
constexpr int CD  = 128;   // model dim
constexpr int CH  = 256;   // hidden
constexpr int CK  = 16;    // energy channels
constexpr int CW  = 9;     // window
constexpr int CJD = 514;   // joint dim (4*D + 2)
constexpr int CJK = 544;   // joint dim padded to mult of 32
constexpr int CTM = 16;    // tokens per workgroup tile

typedef __attribute__((ext_vector_type(16))) _Float16 v16h;
typedef __attribute__((ext_vector_type(8)))  _Float16 h8;
typedef __attribute__((ext_vector_type(8)))  float    v8f;

// ---------------- WMMA helpers (CDNA5 gfx1250, wave32) ----------------
__device__ __forceinline__ v8f wmma_f16(v16h a, v16h b, v8f c) {
  // D = A(16x32 f16) * B(32x16 f16) + C(16x16 f32)
  return __builtin_amdgcn_wmma_f32_16x16x32_f16(false, a, false, b, (short)0, c,
                                                false, false);
}

// A fragment: lane&15 = row M; lane>>4 selects K halves per ISA 16-bit A layout:
//   regs 0..3 hold K = 8h .. 8h+7 ; regs 4..7 hold K = 16+8h .. 16+8h+7
__device__ __forceinline__ v16h frag_a_lds(const _Float16* A, int ldk, int kbase) {
  const int lane = threadIdx.x & 31;
  const int m = lane & 15, h = lane >> 4;
  const h8 lo = *(const h8*)(A + m * ldk + kbase + 8 * h);
  const h8 hi = *(const h8*)(A + m * ldk + kbase + 16 + 8 * h);
  v16h r;
#pragma unroll
  for (int i = 0; i < 8; ++i) { r[i] = lo[i]; r[8 + i] = hi[i]; }
  return r;
}

// B fragment: lane holds column n = nbase + (lane&15); elements 0..7 hold
// K = 8h .. 8h+7, elements 8..15 hold K = 16+8h .. 16+8h+7 (row-major K x N).
__device__ __forceinline__ v16h frag_b_glb(const _Float16* Bm, int ldn, int kbase,
                                           int nbase) {
  const int lane = threadIdx.x & 31;
  const int n = nbase + (lane & 15), h = lane >> 4;
  v16h r;
#pragma unroll
  for (int i = 0; i < 8; ++i) {
    r[i] = Bm[(size_t)(kbase + 8 * h + i) * ldn + n];
    r[8 + i] = Bm[(size_t)(kbase + 16 + 8 * h + i) * ldn + n];
  }
  return r;
}

// Unconditional speculative prefetch of a later K-chunk (global_prefetch_b8).
// Past-the-end addresses land in adjacent ws buffers / are dropped by HW.
__device__ __forceinline__ void prefetch_b(const _Float16* Bm, int ldn, int kbase,
                                           int nbase) {
  const int lane = threadIdx.x & 31;
  const int k = kbase + (lane & 15) + ((lane >> 4) << 4);
  __builtin_prefetch(Bm + (size_t)k * ldn + nbase, 0, 3);
}

__device__ __forceinline__ float gelu_exact(float x) {
  return 0.5f * x * (1.0f + erff(x * 0.70710678118654752f));
}
__device__ __forceinline__ float softplus_f(float x) {
  return (x > 20.f) ? x : log1pf(expf(x));
}

// ---------------- stage 1: pre-LN of states ----------------
__global__ __launch_bounds__(CD) void ln_pre_kernel(const float* __restrict__ states,
                                                    const float* __restrict__ g,
                                                    const float* __restrict__ b,
                                                    float* __restrict__ normG) {
  const size_t t = blockIdx.x;   // token index over B*S
  const int d = threadIdx.x;
  const float x = states[t * CD + d];
  __shared__ float red[8];
  float s = x, s2 = x * x;
#pragma unroll
  for (int off = 16; off; off >>= 1) {
    s += __shfl_xor(s, off, 32);
    s2 += __shfl_xor(s2, off, 32);
  }
  const int wid = threadIdx.x >> 5;
  if ((threadIdx.x & 31) == 0) { red[wid] = s; red[4 + wid] = s2; }
  __syncthreads();
  s = red[0] + red[1] + red[2] + red[3];
  s2 = red[4] + red[5] + red[6] + red[7];
  const float m = s * (1.f / CD);
  const float v = s2 * (1.f / CD) - m * m;
  const float inv = rsqrtf(v + 1e-5f);
  normG[t * CD + d] = (x - m) * inv * g[d] + b[d];
}

// ---------------- stage 0: weight fp32 -> f16 with K padding ----------------
__global__ void cvt_f16_kernel(const float* __restrict__ src, _Float16* __restrict__ dst,
                               int rows, int cols, int padRows) {
  const int idx = blockIdx.x * blockDim.x + threadIdx.x;
  const int total = padRows * cols;
  if (idx >= total) return;
  const int r = idx / cols, c = idx - r * cols;
  dst[idx] = (r < rows) ? (_Float16)src[r * cols + c] : (_Float16)0.f;
}

// ---------------- stage 2: fused per-tile kernel ----------------
__global__ __launch_bounds__(256) void fused_main_kernel(
    const float* __restrict__ states, const float* __restrict__ normG,
    const float* __restrict__ kn_g, const float* __restrict__ kn_b,
    const _Float16* __restrict__ sW1h, const float* __restrict__ sb1,
    const _Float16* __restrict__ sW2h, const float* __restrict__ sb2,
    const _Float16* __restrict__ cW1h, const float* __restrict__ cb1,
    const _Float16* __restrict__ cW2h, const float* __restrict__ cb2,
    const float* __restrict__ e_bias,
    const _Float16* __restrict__ vWh, const float* __restrict__ vb,
    const _Float16* __restrict__ caW1h, const float* __restrict__ cab1,
    const _Float16* __restrict__ caW2h, const float* __restrict__ cab2,
    const _Float16* __restrict__ ocW1h, const float* __restrict__ ocb1,
    const float* __restrict__ ocW2, const float* __restrict__ ocb2,
    const float* __restrict__ out_g, const float* __restrict__ out_b,
    float* __restrict__ out) {
  __shared__ alignas(16) float    normT[24][CD];    // rows s0-8 .. s0+15
  __shared__ alignas(16) _Float16 jointA[CTM][CJK]; // LN'd joint tile (aliased later)
  __shared__ alignas(16) _Float16 hid[CTM][CH];     // hidden activations (f16)
  __shared__ alignas(16) float    acc16[2][CTM][CK];// drive / resistance layer-2
  __shared__ float logitsW[CTM][CW];
  __shared__ float probW[CTM][CW];
  __shared__ float uStr[CTM];
  __shared__ float oldP[CTM];
  __shared__ alignas(16) _Float16 aggH[CTM][CD];    // f16 A-tile scratch
  __shared__ alignas(16) float    msgF[CTM][CD];
  __shared__ alignas(16) float    candF[CTM][CD];

  const int tid = threadIdx.x;
  const int wv = tid >> 5;
  const int lane = tid & 31;
  const int tilesPerSeq = CS / CTM;
  const int bb = blockIdx.x / tilesPerSeq;
  const int s0 = (blockIdx.x % tilesPerSeq) * CTM;

  // load normalized rows [s0-8, s0+16); out-of-range -> zero (matches jnp.pad)
  for (int idx = tid; idx < 24 * CD; idx += 256) {
    const int r = idx >> 7, d = idx & 127;
    const int s = s0 - 8 + r;
    normT[r][d] = (s >= 0) ? normG[((size_t)bb * CS + s) * CD + d] : 0.f;
  }
  __syncthreads();

  // ================= edge loop =================
  for (int w = 0; w < CW; ++w) {
    const int dist = (CW - 1) - w;

    // ---- build LN'd joint tile (wave -> 2 rows); neighbor row = i + w ----
#pragma unroll
    for (int rr = 0; rr < 2; ++rr) {
      const int i = wv * 2 + rr;
      float vbuf[17];
      float s = 0.f, s2 = 0.f;
#pragma unroll
      for (int jj = 0; jj < 17; ++jj) {
        const int j = lane + jj * 32;
        float val = 0.f;
        if (j < CJD) {
          if (j >= 512) {
            val = (j == 512) ? (float)dist * 0.125f : ((dist == 0) ? 1.f : 0.f);
          } else {
            const int d = j & 127, sec = j >> 7;
            const float t = normT[i + 8][d];
            const float n = normT[i + w][d];
            val = (sec == 0) ? t : (sec == 1) ? n : (sec == 2) ? t * n : t - n;
          }
        }
        vbuf[jj] = val;
        s += val; s2 += val * val;
      }
#pragma unroll
      for (int off = 16; off; off >>= 1) {
        s += __shfl_xor(s, off, 32);
        s2 += __shfl_xor(s2, off, 32);
      }
      const float mean = s * (1.f / CJD);
      const float var = s2 * (1.f / CJD) - mean * mean;
      const float inv = rsqrtf(var + 1e-5f);
#pragma unroll
      for (int jj = 0; jj < 17; ++jj) {
        const int j = lane + jj * 32;
        const float outv =
            (j < CJD) ? (vbuf[jj] - mean) * inv * kn_g[j] + kn_b[j] : 0.f;
        jointA[i][j] = (_Float16)outv;   // also zero-pads 514..543
      }
    }
    __syncthreads();

    // ---- drive layer 1: (16x544)@(544x256), wave owns 32 columns ----
    {
      const int n0 = wv * 32;
      v8f a0 = {}, a1 = {};
#pragma clang loop unroll(disable)
      for (int kc = 0; kc < CJK / 32; ++kc) {
        prefetch_b(sW1h, CH, (kc + 2) * 32, n0);
        const v16h af = frag_a_lds(&jointA[0][0], CJK, kc * 32);
        const v16h b0 = frag_b_glb(sW1h, CH, kc * 32, n0);
        const v16h b1 = frag_b_glb(sW1h, CH, kc * 32, n0 + 16);
        a0 = wmma_f16(af, b0, a0);
        a1 = wmma_f16(af, b1, a1);
      }
      const int h = lane >> 4, col = lane & 15;
#pragma unroll
      for (int r = 0; r < 8; ++r) {
        const int row = r + 8 * h;
        hid[row][n0 + col]      = (_Float16)gelu_exact(a0[r] + sb1[n0 + col]);
        hid[row][n0 + 16 + col] = (_Float16)gelu_exact(a1[r] + sb1[n0 + 16 + col]);
      }
    }
    __syncthreads();
    // ---- drive layer 2: (16x256)@(256x16), wave 0, deterministic ----
    if (wv == 0) {
      v8f acc = {};
#pragma clang loop unroll(disable)
      for (int kc = 0; kc < CH / 32; ++kc) {
        const v16h af = frag_a_lds(&hid[0][0], CH, kc * 32);
        const v16h bf = frag_b_glb(sW2h, CK, kc * 32, 0);
        acc = wmma_f16(af, bf, acc);
      }
      const int h = lane >> 4, col = lane & 15;
#pragma unroll
      for (int r = 0; r < 8; ++r) acc16[0][r + 8 * h][col] = acc[r];
    }
    __syncthreads();

    // ---- resistance layer 1 ----
    {
      const int n0 = wv * 32;
      v8f a0 = {}, a1 = {};
#pragma clang loop unroll(disable)
      for (int kc = 0; kc < CJK / 32; ++kc) {
        prefetch_b(cW1h, CH, (kc + 2) * 32, n0);
        const v16h af = frag_a_lds(&jointA[0][0], CJK, kc * 32);
        const v16h b0 = frag_b_glb(cW1h, CH, kc * 32, n0);
        const v16h b1 = frag_b_glb(cW1h, CH, kc * 32, n0 + 16);
        a0 = wmma_f16(af, b0, a0);
        a1 = wmma_f16(af, b1, a1);
      }
      const int h = lane >> 4, col = lane & 15;
#pragma unroll
      for (int r = 0; r < 8; ++r) {
        const int row = r + 8 * h;
        hid[row][n0 + col]      = (_Float16)gelu_exact(a0[r] + cb1[n0 + col]);
        hid[row][n0 + 16 + col] = (_Float16)gelu_exact(a1[r] + cb1[n0 + 16 + col]);
      }
    }
    __syncthreads();
    if (wv == 0) {
      v8f acc = {};
#pragma clang loop unroll(disable)
      for (int kc = 0; kc < CH / 32; ++kc) {
        const v16h af = frag_a_lds(&hid[0][0], CH, kc * 32);
        const v16h bf = frag_b_glb(cW2h, CK, kc * 32, 0);
        acc = wmma_f16(af, bf, acc);
      }
      const int h = lane >> 4, col = lane & 15;
#pragma unroll
      for (int r = 0; r < 8; ++r) acc16[1][r + 8 * h][col] = acc[r];
    }
    __syncthreads();

    // ---- energy -> masked logit (tid = row*16 + k) ----
    {
      const int row = tid >> 4, k = tid & 15;
      const float dr = acc16[0][row][k] + sb2[k];
      const float rs = softplus_f(acc16[1][row][k] + cb2[k]);
      float e = fminf(fmaxf(dr - rs + e_bias[k], -3.f), 3.f);
#pragma unroll
      for (int off = 8; off; off >>= 1) e += __shfl_xor(e, off, 32);
      if (k == 0) {
        const int s = s0 + row;
        logitsW[row][w] = (s >= dist) ? (e * (1.f / CK)) : -1e9f;
      }
    }
    __syncthreads();
  }

  // ---- softmax over W logits + null logit 0 ----
  if (tid < CTM) {
    float mx = 0.f;
#pragma unroll
    for (int w = 0; w < CW; ++w) mx = fmaxf(mx, logitsW[tid][w]);
    float denom = expf(0.f - mx);
    float ex[CW];
#pragma unroll
    for (int w = 0; w < CW; ++w) { ex[w] = expf(logitsW[tid][w] - mx); denom += ex[w]; }
    const float inv = 1.f / denom;
    float u = 0.f;
#pragma unroll
    for (int w = 0; w < CW; ++w) { probW[tid][w] = ex[w] * inv; u += ex[w] * inv; }
    uStr[tid] = u;  // 1 - null weight
  }
  __syncthreads();

  // ---- aggregate neighbors (linearity: message = (sum w*n) @ vW + u*vb) ----
  for (int idx = tid; idx < CTM * CD; idx += 256) {
    const int i = idx >> 7, d = idx & 127;
    float a = 0.f;
#pragma unroll
    for (int w = 0; w < CW; ++w) a += probW[i][w] * normT[i + w][d];
    aggH[i][d] = (_Float16)a;
  }
  __syncthreads();

  // ---- message GEMM: (16x128)@(128x128), wave owns 16 columns ----
  {
    const int n0 = wv * 16;
    v8f acc = {};
#pragma clang loop unroll(disable)
    for (int kc = 0; kc < CD / 32; ++kc) {
      const v16h af = frag_a_lds(&aggH[0][0], CD, kc * 32);
      const v16h bf = frag_b_glb(vWh, CD, kc * 32, n0);
      acc = wmma_f16(af, bf, acc);
    }
    const int h = lane >> 4, col = lane & 15;
#pragma unroll
    for (int r = 0; r < 8; ++r) {
      const int row = r + 8 * h;
      msgF[row][n0 + col] = acc[r] + uStr[row] * vb[n0 + col];
    }
  }
  __syncthreads();

  // ---- candidate input (16x384) aliased into jointA ----
  _Float16(*candIn)[3 * CD] = (_Float16(*)[3 * CD])(&jointA[0][0]);
  for (int idx = tid; idx < CTM * 3 * CD; idx += 256) {
    const int i = idx / (3 * CD), j = idx - i * (3 * CD);
    const int d = j & 127, sec = j >> 7;
    const float nm = normT[i + 8][d], ms = msgF[i][d];
    const float v = (sec == 0) ? nm : (sec == 1) ? ms : nm * ms;
    candIn[i][j] = (_Float16)v;
  }
  __syncthreads();

  // ---- cand layer 1: (16x384)@(384x256) ----
  {
    const int n0 = wv * 32;
    v8f a0 = {}, a1 = {};
#pragma clang loop unroll(disable)
    for (int kc = 0; kc < (3 * CD) / 32; ++kc) {
      prefetch_b(caW1h, CH, (kc + 2) * 32, n0);
      const v16h af = frag_a_lds(&candIn[0][0], 3 * CD, kc * 32);
      const v16h b0 = frag_b_glb(caW1h, CH, kc * 32, n0);
      const v16h b1 = frag_b_glb(caW1h, CH, kc * 32, n0 + 16);
      a0 = wmma_f16(af, b0, a0);
      a1 = wmma_f16(af, b1, a1);
    }
    const int h = lane >> 4, col = lane & 15;
#pragma unroll
    for (int r = 0; r < 8; ++r) {
      const int row = r + 8 * h;
      hid[row][n0 + col]      = (_Float16)gelu_exact(a0[r] + cab1[n0 + col]);
      hid[row][n0 + 16 + col] = (_Float16)gelu_exact(a1[r] + cab1[n0 + 16 + col]);
    }
  }
  __syncthreads();
  // ---- cand layer 2: (16x256)@(256x128) -> tanh ----
  {
    const int n0 = wv * 16;
    v8f acc = {};
#pragma clang loop unroll(disable)
    for (int kc = 0; kc < CH / 32; ++kc) {
      const v16h af = frag_a_lds(&hid[0][0], CH, kc * 32);
      const v16h bf = frag_b_glb(caW2h, CD, kc * 32, n0);
      acc = wmma_f16(af, bf, acc);
    }
    const int h = lane >> 4, col = lane & 15;
#pragma unroll
    for (int r = 0; r < 8; ++r) {
      const int row = r + 8 * h;
      candF[row][n0 + col] = tanhf(acc[r] + cab2[n0 + col]);
    }
  }
  __syncthreads();

  // ---- old precision layer 1: (16x128)@(128x256) ----
  for (int idx = tid; idx < CTM * CD; idx += 256) {
    const int i = idx >> 7, d = idx & 127;
    aggH[i][d] = (_Float16)normT[i + 8][d];
  }
  __syncthreads();
  {
    const int n0 = wv * 32;
    v8f a0 = {}, a1 = {};
#pragma clang loop unroll(disable)
    for (int kc = 0; kc < CD / 32; ++kc) {
      const v16h af = frag_a_lds(&aggH[0][0], CD, kc * 32);
      const v16h b0 = frag_b_glb(ocW1h, CH, kc * 32, n0);
      const v16h b1 = frag_b_glb(ocW1h, CH, kc * 32, n0 + 16);
      a0 = wmma_f16(af, b0, a0);
      a1 = wmma_f16(af, b1, a1);
    }
    const int h = lane >> 4, col = lane & 15;
#pragma unroll
    for (int r = 0; r < 8; ++r) {
      const int row = r + 8 * h;
      hid[row][n0 + col]      = (_Float16)gelu_exact(a0[r] + ocb1[n0 + col]);
      hid[row][n0 + 16 + col] = (_Float16)gelu_exact(a1[r] + ocb1[n0 + 16 + col]);
    }
  }
  __syncthreads();
  // ---- old precision layer 2: N=1 dot + softplus ----
  {
    const int row = tid >> 4, k16 = tid & 15;
    float s = 0.f;
#pragma unroll
    for (int q = 0; q < 16; ++q) {
      const int hc = k16 * 16 + q;
      s += (float)hid[row][hc] * ocW2[hc];
    }
#pragma unroll
    for (int off = 8; off; off >>= 1) s += __shfl_xor(s, off, 32);
    if (k16 == 0) oldP[row] = softplus_f(s + ocb2[0]);
  }
  __syncthreads();

  // ---- precision-weighted update + output LN (wave -> 2 rows) ----
#pragma unroll
  for (int rr = 0; rr < 2; ++rr) {
    const int i = wv * 2 + rr;
    const int srow = s0 + i;
    const float* st = states + ((size_t)bb * CS + srow) * CD;
    const float u = uStr[i], op = oldP[i];
    const float invden = 1.f / (op + u + 1e-6f);
    float vals[4];
    float s = 0.f, s2 = 0.f;
#pragma unroll
    for (int q = 0; q < 4; ++q) {
      const int d = lane * 4 + q;
      const float upd = (op * st[d] + u * candF[i][d]) * invden;
      vals[q] = upd; s += upd; s2 += upd * upd;
    }
#pragma unroll
    for (int off = 16; off; off >>= 1) {
      s += __shfl_xor(s, off, 32);
      s2 += __shfl_xor(s2, off, 32);
    }
    const float mean = s * (1.f / CD);
    const float var = s2 * (1.f / CD) - mean * mean;
    const float inv = rsqrtf(var + 1e-5f);
    float* po = out + ((size_t)bb * CS + srow) * CD;
#pragma unroll
    for (int q = 0; q < 4; ++q) {
      const int d = lane * 4 + q;
      po[d] = (vals[q] - mean) * inv * out_g[d] + out_b[d];
    }
  }
}

// ---------------- launcher ----------------
extern "C" void kernel_launch(void* const* d_in, const int* in_sizes, int n_in,
                              void* d_out, int out_size, void* d_ws, size_t ws_size,
                              hipStream_t stream) {
  (void)in_sizes; (void)n_in; (void)out_size; (void)ws_size;
  const float* states = (const float*)d_in[0];
  const float* pre_g  = (const float*)d_in[1];
  const float* pre_b  = (const float*)d_in[2];
  const float* kn_g   = (const float*)d_in[3];
  const float* kn_b   = (const float*)d_in[4];
  const float* sW1    = (const float*)d_in[5];
  const float* sb1    = (const float*)d_in[6];
  const float* sW2    = (const float*)d_in[7];
  const float* sb2    = (const float*)d_in[8];
  const float* cW1    = (const float*)d_in[9];
  const float* cb1    = (const float*)d_in[10];
  const float* cW2    = (const float*)d_in[11];
  const float* cb2    = (const float*)d_in[12];
  const float* e_bias = (const float*)d_in[13];
  const float* vW     = (const float*)d_in[14];
  const float* vb     = (const float*)d_in[15];
  const float* caW1   = (const float*)d_in[16];
  const float* cab1   = (const float*)d_in[17];
  const float* caW2   = (const float*)d_in[18];
  const float* cab2   = (const float*)d_in[19];
  const float* ocW1   = (const float*)d_in[20];
  const float* ocb1   = (const float*)d_in[21];
  const float* ocW2   = (const float*)d_in[22];
  const float* ocb2   = (const float*)d_in[23];
  const float* out_g  = (const float*)d_in[24];
  const float* out_b  = (const float*)d_in[25];
  float* out = (float*)d_out;

  char* ws = (char*)d_ws;
  size_t off = 0;
  auto walloc = [&](size_t bytes) -> char* {
    char* p = ws + off;
    off += (bytes + 255) & ~(size_t)255;
    return p;
  };
  float*     normG = (float*)walloc((size_t)CB * CS * CD * sizeof(float));
  _Float16*  sW1h  = (_Float16*)walloc((size_t)CJK * CH * 2);
  _Float16*  cW1h  = (_Float16*)walloc((size_t)CJK * CH * 2);
  _Float16*  sW2h  = (_Float16*)walloc((size_t)CH * CK * 2);
  _Float16*  cW2h  = (_Float16*)walloc((size_t)CH * CK * 2);
  _Float16*  vWh   = (_Float16*)walloc((size_t)CD * CD * 2);
  _Float16*  caW1h = (_Float16*)walloc((size_t)3 * CD * CH * 2);
  _Float16*  caW2h = (_Float16*)walloc((size_t)CH * CD * 2);
  _Float16*  ocW1h = (_Float16*)walloc((size_t)CD * CH * 2);

  // stage 0: weight conversion (K padded to mult of 32)
  auto cvt = [&](const float* src, _Float16* dst, int rows, int cols, int padRows) {
    const int total = padRows * cols;
    hipLaunchKernelGGL(cvt_f16_kernel, dim3((total + 255) / 256), dim3(256), 0, stream,
                       src, dst, rows, cols, padRows);
  };
  cvt(sW1,  sW1h,  CJD,    CH, CJK);
  cvt(cW1,  cW1h,  CJD,    CH, CJK);
  cvt(sW2,  sW2h,  CH,     CK, CH);
  cvt(cW2,  cW2h,  CH,     CK, CH);
  cvt(vW,   vWh,   CD,     CD, CD);
  cvt(caW1, caW1h, 3 * CD, CH, 3 * CD);
  cvt(caW2, caW2h, CH,     CD, CH);
  cvt(ocW1, ocW1h, CD,     CH, CD);

  // stage 1: pre-LN
  hipLaunchKernelGGL(ln_pre_kernel, dim3(CB * CS), dim3(CD), 0, stream,
                     states, pre_g, pre_b, normG);

  // stage 2: fused tile kernel
  hipLaunchKernelGGL(fused_main_kernel, dim3((CB * CS) / CTM), dim3(256), 0, stream,
                     states, normG, kn_g, kn_b, sW1h, sb1, sW2h, sb2, cW1h, cb1,
                     cW2h, cb2, e_bias, vWh, vb, caW1h, cab1, caW2h, cab2, ocW1h,
                     ocb1, ocW2, ocb2, out_g, out_b, out);
}
